// FAModule_13881334300957
// MI455X (gfx1250) — matrix operand order repaired and checked by reading, not compiled, not run on hardware
//
#include <hip/hip_runtime.h>
#include <hip/hip_bf16.h>
#include <stdint.h>

typedef __bf16 v16bf __attribute__((ext_vector_type(16)));
typedef float  v8f   __attribute__((ext_vector_type(8)));
typedef float  v4f   __attribute__((ext_vector_type(4)));
typedef unsigned int v4u   __attribute__((ext_vector_type(4)));
typedef unsigned int u32x4 __attribute__((ext_vector_type(4)));
typedef int          i32x4 __attribute__((ext_vector_type(4)));
typedef int          i32x8 __attribute__((ext_vector_type(8)));

#ifndef __has_builtin
#define __has_builtin(x) 0
#endif
#if __has_builtin(__builtin_amdgcn_tensor_load_to_lds) && \
    __has_builtin(__builtin_amdgcn_s_wait_tensorcnt)
#define HAVE_TDM 1
#else
#define HAVE_TDM 0
#endif

__device__ __forceinline__ unsigned short f2bf(float f) {
  union { float f; unsigned int u; } c; c.f = f;
  unsigned int u = c.u;
  unsigned int r = 0x7FFFu + ((u >> 16) & 1u);   // round-to-nearest-even
  return (unsigned short)((u + r) >> 16);
}

#if HAVE_TDM
// 2D f32 tile (tile0 x tile1) from row-major tensor (stride0 elems/row) -> LDS.
// D# layout per CDNA5 ISA:
//   group0: count=1, lds_addr[63:32], global_addr[120:64], type=2 [127:126]
//   group1: data_size=2(4B)[17:16], tensor_dim0[79:48], tensor_dim1[111:80],
//           tile_dim0[127:112], tile_dim1[143:128], tensor_dim0_stride[207:160]
// OOB reads (beyond tensor_dim0/1) return zero -> free edge handling.
__device__ __forceinline__ void tdm_load_f32_2d(const float* gaddr, unsigned lds_off,
                                                unsigned tile0, unsigned tile1,
                                                unsigned tdim0, unsigned tdim1,
                                                unsigned stride0) {
  union { unsigned long long q[2]; u32x4 v; } g0;
  union { unsigned long long q[4]; i32x8 v; } g1;
  union { unsigned long long q[2]; i32x4 v; } gz;
  unsigned long long ga = (unsigned long long)(uintptr_t)gaddr;
  g0.q[0] = 1ull | ((unsigned long long)lds_off << 32);
  g0.q[1] = (ga & ((1ull << 57) - 1ull)) | (2ull << 62);
  g1.q[0] = (2ull << 16) | (((unsigned long long)(tdim0 & 0xffffu)) << 48);
  g1.q[1] = ((unsigned long long)(tdim0 >> 16) & 0xffffu)
          | (((unsigned long long)tdim1 & 0xffffffffull) << 16)
          | (((unsigned long long)(tile0 & 0xffffu)) << 48);
  g1.q[2] = ((unsigned long long)(tile1 & 0xffffu))
          | (((unsigned long long)stride0 & 0xffffffffull) << 32);
  g1.q[3] = 0ull; // stride0[47:32]=0 (small), tensor_dim1_stride unused (2D)
  gz.q[0] = 0; gz.q[1] = 0;
#if defined(__clang_major__) && __clang_major__ >= 23
  i32x8 z8 = {0, 0, 0, 0, 0, 0, 0, 0};
  __builtin_amdgcn_tensor_load_to_lds(g0.v, g1.v, gz.v, gz.v, z8, 0);
#else
  __builtin_amdgcn_tensor_load_to_lds(g0.v, g1.v, gz.v, gz.v, 0);
#endif
}
#endif

union frag16 { v16bf v; unsigned short u[16]; v4u q[2]; };

// ---------------------------------------------------------------------------
// Implicit-GEMM conv via V_WMMA_F32_16X16X32_BF16.
// Grid = (ceil(H*W/256), B): N tile never crosses a batch, so for 1x1 convs
// the B operand is a rectangular 2D tile staged by the Tensor Data Mover
// (TDM -> LDS -> WMMA). 3x3 convs use an incremental im2col gather with
// packed bf16 staged via 128-bit LDS ops (col-major, 40-ushort pitch).
// 256 threads = 8 waves; each wave owns a 32x32 output tile => 4 WMMA / chunk.
// ---------------------------------------------------------------------------
__global__ __launch_bounds__(256) void k_conv_wmma(
    const float* __restrict__ X, const float* __restrict__ W,
    const float* __restrict__ bias, const float* __restrict__ resid,
    float* __restrict__ Y,
    int Cin, int Cout, int H, int Wd, int ks, int pad, int relu)
{
  __shared__ __attribute__((aligned(16))) float Af[32 * 32];   // weights (f32; TDM)
  __shared__ __attribute__((aligned(16))) float Bf[32 * 256];  // acts: f32 TDM view
  unsigned short* Bsu = (unsigned short*)Bf;  // gather view: [col][40] bf16 pitch

  const int tid  = threadIdx.x;
  const int b    = blockIdx.y;
  const int hw0  = blockIdx.x * 256;
  const int HWsz = H * Wd;
  const int ks2  = ks * ks;
  const int KK   = Cin * ks2;
  const int nK   = (KK + 31) >> 5;

  const int lane = tid & 31;
  const int g    = lane >> 4;
  const int mrow = lane & 15;
  const int jA   = (tid >> 5) * 32;             // wave's N sub-tile base (2x16)

  const int hw_t = hw0 + tid;                   // this thread's B column
  const bool nv  = hw_t < HWsz;
  const float* Xb = X + (size_t)b * Cin * HWsz;

#if HAVE_TDM
  const bool tdmB = (ks == 1);
  const unsigned ldsAf = (unsigned)(unsigned long long)(uintptr_t)Af;
  const unsigned ldsBf = (unsigned)(unsigned long long)(uintptr_t)Bf;
#else
  const bool tdmB = false;
#endif

  int py = 0, px = 0;                           // only needed by gather path
  if (!tdmB && nv) { py = hw_t / Wd; px = hw_t - py * Wd; }

  v8f acc00 = {0.f,0.f,0.f,0.f,0.f,0.f,0.f,0.f};
  v8f acc01 = {0.f,0.f,0.f,0.f,0.f,0.f,0.f,0.f};
  v8f acc10 = {0.f,0.f,0.f,0.f,0.f,0.f,0.f,0.f};
  v8f acc11 = {0.f,0.f,0.f,0.f,0.f,0.f,0.f,0.f};

  for (int kc = 0; kc < nK; ++kc) {
    const int kbase = kc << 5;

    // ---- stage A chunk [Cout x 32] (and, for 1x1, B chunk [32 x 256]) ----
#if HAVE_TDM
    if (tid < 32) {
      tdm_load_f32_2d(W + kbase, ldsAf, /*tile0=*/32, /*tile1=*/(unsigned)Cout,
                      /*tdim0=*/(unsigned)(KK - kbase), /*tdim1=*/(unsigned)Cout,
                      /*stride0=*/(unsigned)KK);
      if (tdmB)
        tdm_load_f32_2d(Xb + (size_t)kbase * HWsz + hw0, ldsBf,
                        /*tile0=*/256, /*tile1=*/32,
                        /*tdim0=*/(unsigned)(HWsz - hw0),
                        /*tdim1=*/(unsigned)(Cin - kbase),
                        /*stride0=*/(unsigned)HWsz);
      __builtin_amdgcn_s_wait_tensorcnt(0);
    }
#else
    for (int idx = tid; idx < Cout * 32; idx += 256) {
      int m = idx >> 5, kk = idx & 31, k = kbase + kk;
      Af[idx] = (k < KK) ? W[(size_t)m * KK + k] : 0.f;
    }
#endif

    // ---- gather path (3x3): pack bf16 pairs in regs, 4x ds_store_b128 ----
    if (!tdmB) {
      if (nv && kc + 1 < nK) {                  // hint next chunk into caches
        int cn = (kbase + 32) / ks2;
        __builtin_prefetch(Xb + (size_t)cn * HWsz + hw_t, 0, 1);
      }
      int c  = kbase / ks2;
      int kt = kbase - c * ks2;
      int ky = kt / ks, kx = kt - ky * ks;
      unsigned int pk[16];
#pragma unroll
      for (int w = 0; w < 16; ++w) pk[w] = 0u;
#pragma unroll
      for (int it = 0; it < 32; ++it) {
        float v = 0.f;
        if (nv && (kbase + it) < KK) {
          int yy = py + ky - pad, xx = px + kx - pad;
          if (yy >= 0 && yy < H && xx >= 0 && xx < Wd)
            v = Xb[(size_t)c * HWsz + yy * Wd + xx];
        }
        pk[it >> 1] |= (unsigned int)f2bf(v) << ((it & 1) * 16);
        if (++kx == ks) { kx = 0; if (++ky == ks) { ky = 0; ++c; } }
      }
      v4u* dst = (v4u*)&Bsu[tid * 40];          // 80B pitch: 16B-aligned
#pragma unroll
      for (int t = 0; t < 4; ++t) {
        v4u qv = {pk[4 * t], pk[4 * t + 1], pk[4 * t + 2], pk[4 * t + 3]};
        dst[t] = qv;
      }
    }
    __syncthreads();

    // ---- fragments (contiguous k order: element s holds k = 16*g + s) + 4 WMMAs ----
    frag16 fa0, fa1, fb0, fb1;
    {
      // A rows: two contiguous 8-float runs per row -> 4x ds_load_b128 each
      const v4f* r0a = (const v4f*)&Af[mrow * 32 + 16 * g];
      const v4f* r0b = (const v4f*)&Af[mrow * 32 + 8 + 16 * g];
      const v4f* r1a = (const v4f*)&Af[(16 + mrow) * 32 + 16 * g];
      const v4f* r1b = (const v4f*)&Af[(16 + mrow) * 32 + 8 + 16 * g];
      v4f a0 = r0a[0], a1 = r0a[1], a2 = r0b[0], a3 = r0b[1];
      v4f b0 = r1a[0], b1 = r1a[1], b2 = r1b[0], b3 = r1b[1];
#pragma unroll
      for (int i = 0; i < 4; ++i) {
        fa0.u[i]      = f2bf(a0[i]);
        fa0.u[4 + i]  = f2bf(a1[i]);
        fa0.u[8 + i]  = f2bf(a2[i]);
        fa0.u[12 + i] = f2bf(a3[i]);
        fa1.u[i]      = f2bf(b0[i]);
        fa1.u[4 + i]  = f2bf(b1[i]);
        fa1.u[8 + i]  = f2bf(b2[i]);
        fa1.u[12 + i] = f2bf(b3[i]);
      }
    }
    if (tdmB) {
#pragma unroll
      for (int i = 0; i < 16; ++i) {            // TDM tile is [k][256] row-major
        int rB = (i + 16 * g) * 256;
        fb0.u[i] = f2bf(Bf[rB + jA + mrow]);
        fb1.u[i] = f2bf(Bf[rB + jA + 16 + mrow]);
      }
    } else {
      // col-major gather tile: 16 contiguous bf16 per lane -> 2x ds_load_b128
      const v4u* c0 = (const v4u*)&Bsu[(jA + mrow) * 40 + 16 * g];
      const v4u* c1 = (const v4u*)&Bsu[(jA + 16 + mrow) * 40 + 16 * g];
      fb0.q[0] = c0[0]; fb0.q[1] = c0[1];
      fb1.q[0] = c1[0]; fb1.q[1] = c1[1];
    }
    acc00 = __builtin_amdgcn_wmma_f32_16x16x32_bf16(false, fa0.v, false, fb0.v,
                                                    (short)0, acc00, false, false);
    acc01 = __builtin_amdgcn_wmma_f32_16x16x32_bf16(false, fa0.v, false, fb1.v,
                                                    (short)0, acc01, false, false);
    if (Cout > 16) {
      acc10 = __builtin_amdgcn_wmma_f32_16x16x32_bf16(false, fa1.v, false, fb0.v,
                                                      (short)0, acc10, false, false);
      acc11 = __builtin_amdgcn_wmma_f32_16x16x32_bf16(false, fa1.v, false, fb1.v,
                                                      (short)0, acc11, false, false);
    }
    __syncthreads();
  }

  // ---- epilogue: bias / relu / residual, f32 stores (C/D 16x16 layout) ----
#pragma unroll
  for (int half = 0; half < 2; ++half) {
    int hw = hw0 + jA + half * 16 + mrow;
    if (hw < HWsz) {
      v8f a0 = half ? acc01 : acc00;
      v8f a1 = half ? acc11 : acc10;
#pragma unroll
      for (int r = 0; r < 8; ++r) {
        int M0 = r + 8 * g;
        {
          float v = a0[r];
          if (bias) v += bias[M0];
          if (relu) v = fmaxf(v, 0.f);
          size_t oidx = ((size_t)(b * Cout + M0)) * HWsz + hw;
          if (resid) v += resid[oidx];
          Y[oidx] = v;
        }
        if (Cout > 16) {
          int M1 = 16 + M0;
          float v = a1[r];
          if (bias) v += bias[M1];
          if (relu) v = fmaxf(v, 0.f);
          size_t oidx = ((size_t)(b * Cout + M1)) * HWsz + hw;
          if (resid) v += resid[oidx];
          Y[oidx] = v;
        }
      }
    }
  }
}

// ---------------------------------------------------------------------------
// Elementwise / glue kernels
// ---------------------------------------------------------------------------
__global__ void k_resize(const float* __restrict__ X, float* __restrict__ Y,
                         int Bn, int C, int Hin, int Win, int Hout, int Wout,
                         int dstCtot, int dstCoff)
{
  int idx = blockIdx.x * blockDim.x + threadIdx.x;
  int total = Bn * C * Hout * Wout;
  if (idx >= total) return;
  int x = idx % Wout; int t = idx / Wout;
  int y = t % Hout;   t /= Hout;
  int c = t % C;      int b = t / C;
  float sy = (Hout > 1) ? y * (float)(Hin - 1) / (float)(Hout - 1) : 0.f;
  float sx = (Wout > 1) ? x * (float)(Win - 1) / (float)(Wout - 1) : 0.f;
  int y0 = (int)sy; if (y0 > Hin - 1) y0 = Hin - 1;
  int x0 = (int)sx; if (x0 > Win - 1) x0 = Win - 1;
  float fy = sy - y0, fx = sx - x0;
  int y1 = (y0 + 1 < Hin) ? y0 + 1 : Hin - 1;
  int x1 = (x0 + 1 < Win) ? x0 + 1 : Win - 1;
  const float* p = X + (size_t)(b * C + c) * Hin * Win;
  float v = (1.f - fy) * ((1.f - fx) * p[y0 * Win + x0] + fx * p[y0 * Win + x1]) +
            fy * ((1.f - fx) * p[y1 * Win + x0] + fx * p[y1 * Win + x1]);
  Y[(((size_t)b * dstCtot + dstCoff + c) * Hout + y) * Wout + x] = v;
}

__global__ void k_absdiff(const float* A, const float* B, float* Y, int n) {
  int i = blockIdx.x * blockDim.x + threadIdx.x;
  if (i < n) Y[i] = fabsf(A[i] - B[i]);
}
__global__ void k_add3(const float* A, const float* B, const float* C, float* Y, int n) {
  int i = blockIdx.x * blockDim.x + threadIdx.x;
  if (i < n) Y[i] = A[i] + B[i] + C[i];
}
__global__ void k_mul3_off(const float* A, const float* B, const float* C, float* Y,
                           int Bn, int Cc, int HW, int dstCtot, int dstCoff) {
  int i = blockIdx.x * blockDim.x + threadIdx.x;
  int total = Bn * Cc * HW;
  if (i >= total) return;
  int hw = i % HW; int t = i / HW;
  int c = t % Cc;  int b = t / Cc;
  float v = A[i] * B[i] * C[i];
  Y[(((size_t)b * dstCtot + dstCoff + c) * HW) + hw] = v;
}
__global__ void k_bafuse(const float* a, const float* m, const float* p,
                         const float* q, const float* r, float* Y, int n) {
  int i = blockIdx.x * blockDim.x + threadIdx.x;
  if (i < n) { float mm = m[i]; Y[i] = a[i] * mm * mm + p[i] + q[i] + r[i]; }
}

__global__ __launch_bounds__(256) void k_se_reduce(const float* __restrict__ X,
                                                   float* avg, float* mx, int HW) {
  int bc = blockIdx.x;
  const float* p = X + (size_t)bc * HW;
  float s = 0.f, m = -1e30f;
  for (int i = threadIdx.x; i < HW; i += 256) { float v = p[i]; s += v; m = fmaxf(m, v); }
  __shared__ float ss[256], sm[256];
  ss[threadIdx.x] = s; sm[threadIdx.x] = m; __syncthreads();
  for (int off = 128; off > 0; off >>= 1) {
    if (threadIdx.x < off) {
      ss[threadIdx.x] += ss[threadIdx.x + off];
      sm[threadIdx.x] = fmaxf(sm[threadIdx.x], sm[threadIdx.x + off]);
    }
    __syncthreads();
  }
  if (threadIdx.x == 0) { avg[bc] = ss[0] / (float)HW; mx[bc] = sm[0]; }
}

__global__ void k_se_attn(const float* avg, const float* mx,
                          const float* wfc1, const float* wfc2, float* attn) {
  int b = blockIdx.x, c = threadIdx.x;
  __shared__ float av[64], mv[64], h[8];
  av[c] = avg[b * 64 + c]; mv[c] = mx[b * 64 + c];
  __syncthreads();
  if (c < 8) {
    int j = c & 3; const float* src = (c < 4) ? av : mv;
    float s = 0.f;
    for (int k = 0; k < 64; ++k) s += wfc1[j * 64 + k] * src[k];
    h[c] = fmaxf(s, 0.f);
  }
  __syncthreads();
  float s = 0.f;
  for (int j = 0; j < 4; ++j) s += wfc2[c * 4 + j] * (h[j] + h[4 + j]);
  attn[b * 64 + c] = 1.f / (1.f + __expf(-s));
}

__global__ void k_spat_mm(const float* __restrict__ x1, const float* __restrict__ attn,
                          float* __restrict__ sa2, int HW) {
  int idx = blockIdx.x * blockDim.x + threadIdx.x;
  if (idx >= 32 * HW) return;
  int b = idx / HW, hw = idx - b * HW;
  float s = 0.f, m = -1e30f;
  for (int c = 0; c < 64; ++c) {
    float v = attn[b * 64 + c] * x1[((size_t)(b * 64 + c)) * HW + hw];
    s += v; m = fmaxf(m, v);
  }
  sa2[((size_t)b * 2 + 0) * HW + hw] = s / 64.f;
  sa2[((size_t)b * 2 + 1) * HW + hw] = m;
}

__global__ void k_conv7_sig(const float* __restrict__ sa2, const float* __restrict__ wsa,
                            float* __restrict__ spat, int H, int W) {
  int idx = blockIdx.x * blockDim.x + threadIdx.x;
  if (idx >= 32 * H * W) return;
  int b = idx / (H * W), hw = idx - b * H * W;
  int y = hw / W, x = hw - y * W;
  float s = 0.f;
  for (int ci = 0; ci < 2; ++ci)
    for (int ky = 0; ky < 7; ++ky) {
      int yy = y + ky - 3; if (yy < 0 || yy >= H) continue;
      for (int kx = 0; kx < 7; ++kx) {
        int xx = x + kx - 3; if (xx < 0 || xx >= W) continue;
        s += wsa[(ci * 7 + ky) * 7 + kx] * sa2[((size_t)(b * 2 + ci) * H + yy) * W + xx];
      }
    }
  spat[idx] = 1.f / (1.f + __expf(-s));
}

__global__ void k_ta(const float* x1, const float* attn, const float* spat,
                     float* ta, int HW) {
  int i = blockIdx.x * blockDim.x + threadIdx.x;
  if (i >= 32 * 64 * HW) return;
  int hw = i % HW; int t = i / HW;
  int c = t & 63;  int b = t >> 6;
  ta[i] = spat[b * HW + hw] * attn[b * 64 + c] * x1[i];
}

__global__ void k_edge(const float* __restrict__ sa, float* __restrict__ edge, int HW) {
  int idx = blockIdx.x * blockDim.x + threadIdx.x;
  if (idx >= 32 * HW) return;
  int b = idx / HW, hw = idx - b * HW;
  const float* p = sa + (size_t)b * 32 * HW + hw;
  float mx = -1e30f;
#pragma unroll
  for (int c = 0; c < 32; ++c) mx = fmaxf(mx, p[(size_t)c * HW]);
  float s = 0.f, v1 = 0.f;
#pragma unroll
  for (int c = 0; c < 32; ++c) {
    float e = __expf(p[(size_t)c * HW] - mx);
    s += e; if (c == 1) v1 = e;
  }
  edge[idx] = v1 / s;
}

__global__ void k_xmask(const float* xp, const float* edge, float* xm, int HW) {
  int i = blockIdx.x * blockDim.x + threadIdx.x;
  if (i >= 32 * 16 * HW) return;
  int hw = i % HW; int b = i / (16 * HW);
  xm[i] = xp[i] * edge[b * HW + hw];
}

__global__ void k_anchor(const float* __restrict__ xm, float* __restrict__ anchor) {
  int idx = blockIdx.x * blockDim.x + threadIdx.x;
  if (idx >= 32 * 16 * 16) return;
  int n = idx & 15; int s = (idx >> 4) & 15; int b = idx >> 8;
  int pi = (n >> 2) + 2, pj = (n & 3) + 2;
  int r0 = (pi * 44) / 8, r1 = ((pi + 1) * 44 + 7) / 8;
  int c0 = (pj * 44) / 8, c1 = ((pj + 1) * 44 + 7) / 8;
  float sum = 0.f;
  const float* p = xm + (size_t)(b * 16 + s) * 44 * 44;
  for (int r = r0; r < r1; ++r)
    for (int c = c0; c < c1; ++c) sum += p[r * 44 + c];
  anchor[idx] = sum / (float)((r1 - r0) * (c1 - c0));
}

__global__ void k_proj(const float* __restrict__ anchor, const float* __restrict__ xp,
                       float* __restrict__ proj, int HW) {
  int idx = blockIdx.x * blockDim.x + threadIdx.x;
  if (idx >= 32 * HW) return;
  int b = idx / HW, L = idx - b * HW;
  float xv[16];
#pragma unroll
  for (int s = 0; s < 16; ++s) xv[s] = xp[((size_t)(b * 16 + s)) * HW + L];
  float v[16]; float mx = -1e30f;
#pragma unroll
  for (int n = 0; n < 16; ++n) {
    float s = 0.f;
#pragma unroll
    for (int ss = 0; ss < 16; ++ss) s += anchor[(b * 16 + ss) * 16 + n] * xv[ss];
    v[n] = s; mx = fmaxf(mx, s);
  }
  float se = 0.f;
#pragma unroll
  for (int n = 0; n < 16; ++n) { v[n] = __expf(v[n] - mx); se += v[n]; }
  float inv = 1.f / se;
#pragma unroll
  for (int n = 0; n < 16; ++n) proj[((size_t)(b * 16 + n)) * HW + L] = v[n] * inv;
}

__global__ __launch_bounds__(256) void k_xb(const float* __restrict__ xstate,
                                            const float* __restrict__ proj,
                                            float* __restrict__ xb, int HW) {
  int b = blockIdx.x >> 4, n = blockIdx.x & 15;
  float acc[16];
#pragma unroll
  for (int s = 0; s < 16; ++s) acc[s] = 0.f;
  for (int L = threadIdx.x; L < HW; L += 256) {
    float pv = proj[((size_t)(b * 16 + n)) * HW + L];
#pragma unroll
    for (int s = 0; s < 16; ++s) acc[s] += xstate[((size_t)(b * 16 + s)) * HW + L] * pv;
  }
  __shared__ float red[256];
#pragma unroll
  for (int s = 0; s < 16; ++s) {
    red[threadIdx.x] = acc[s]; __syncthreads();
    for (int off = 128; off > 0; off >>= 1) {
      if (threadIdx.x < off) red[threadIdx.x] += red[threadIdx.x + off];
      __syncthreads();
    }
    if (threadIdx.x == 0) xb[(b * 16 + s) * 16 + n] = red[0];
    __syncthreads();
  }
}

__global__ void k_gcn(const float* xb, const float* wg1, const float* bg1,
                      const float* wg2, float* xb2) {
  int b = blockIdx.x, t = threadIdx.x;
  __shared__ float xs_[256], h[256];
  xs_[t] = xb[b * 256 + t];
  __syncthreads();
  int s = t >> 4, i = t & 15;
  float acc = 0.f;
  for (int j = 0; j < 16; ++j) acc += wg1[i * 16 + j] * xs_[s * 16 + j];
  acc += bg1[i] + xs_[s * 16 + i];
  h[s * 16 + i] = fmaxf(acc, 0.f);
  __syncthreads();
  float o = 0.f;
  for (int j = 0; j < 16; ++j) o += wg2[(t >> 4) * 16 + j] * h[j * 16 + (t & 15)];
  xb2[b * 256 + t] = o;
}

__global__ void k_xs(const float* __restrict__ xb2, const float* __restrict__ proj,
                     float* __restrict__ xs, int HW) {
  int idx = blockIdx.x * blockDim.x + threadIdx.x;
  if (idx >= 32 * HW) return;
  int b = idx / HW, L = idx - b * HW;
  float pv[16];
#pragma unroll
  for (int n = 0; n < 16; ++n) pv[n] = proj[((size_t)(b * 16 + n)) * HW + L];
#pragma unroll
  for (int s = 0; s < 16; ++s) {
    float a = 0.f;
#pragma unroll
    for (int n = 0; n < 16; ++n) a += xb2[(b * 16 + s) * 16 + n] * pv[n];
    xs[((size_t)(b * 16 + s)) * HW + L] = a;
  }
}

// ---------------------------------------------------------------------------
// Host side
// ---------------------------------------------------------------------------
struct P {
  const float *w2,*w3,*w4,*ws32,*ws42,*ws43,*ws432,*wm32,*wm42,*wm43,*w5,
              *wfc1,*wfc2,*wsa,*w1,*wst,*wpr,*wex;
  const float *b2,*b3,*b4,*bs32,*bs42,*bs43,*bs432,*bm32,*bm42,*bm43,*b5,*b1,*bst,*bpr;
  const float *wg1,*bg1,*wg2;
  const float *x1,*x2,*x3,*x4;
};

static inline void launch_conv(hipStream_t st, const float* X, const float* Wt,
                               const float* bias, const float* resid, float* Y,
                               int Bn, int Cin, int Cout, int H, int Wd,
                               int ks, int pad, int relu) {
  int bpb = (H * Wd + 255) / 256;           // N tiles per batch image
  dim3 grid(bpb, Bn);
  k_conv_wmma<<<grid, 256, 0, st>>>(X, Wt, bias, resid, Y, Cin, Cout, H, Wd, ks, pad, relu);
}

extern "C" void kernel_launch(void* const* d_in, const int* in_sizes, int n_in,
                              void* d_out, int out_size, void* d_ws, size_t ws_size,
                              hipStream_t stream) {
  if (n_in < 39) return;
  P p;
  auto f = [&](int i) { return (const float*)d_in[i]; };

  const int X1SZ = 32 * 64 * 88 * 88;  // 15,859,712 : unique, identifies layout
  if (in_sizes[0] == X1SZ) {
    // insertion order: x1..x4 then params in dict-insertion order
    p.x1=f(0); p.x2=f(1); p.x3=f(2); p.x4=f(3);
    p.w2=f(4); p.w3=f(5); p.w4=f(6); p.ws32=f(7); p.ws42=f(8); p.ws43=f(9);
    p.ws432=f(10); p.wm32=f(11); p.wm42=f(12); p.wm43=f(13); p.w5=f(14);
    p.wfc1=f(15); p.wfc2=f(16); p.wsa=f(17); p.w1=f(18); p.wst=f(19);
    p.wpr=f(20); p.wex=f(21);
    p.b2=f(22); p.b3=f(23); p.b4=f(24); p.bs32=f(25); p.bs42=f(26); p.bs43=f(27);
    p.bs432=f(28); p.bm32=f(29); p.bm42=f(30); p.bm43=f(31); p.b5=f(32);
    p.b1=f(33); p.bst=f(34); p.bpr=f(35);
    p.wg1=f(36); p.bg1=f(37); p.wg2=f(38);
  } else {
    // JAX-sorted pytree: params (alphabetical) then x1..x4
    p.b1=f(0); p.b2=f(1); p.b3=f(2); p.b4=f(3); p.b5=f(4); p.bg1=f(5);
    p.bm32=f(6); p.bm42=f(7); p.bm43=f(8); p.bpr=f(9); p.bs32=f(10);
    p.bs42=f(11); p.bs43=f(12); p.bs432=f(13); p.bst=f(14);
    p.w1=f(15); p.w2=f(16); p.w3=f(17); p.w4=f(18); p.w5=f(19); p.wex=f(20);
    p.wfc1=f(21); p.wfc2=f(22); p.wg1=f(23); p.wg2=f(24);
    p.wm32=f(25); p.wm42=f(26); p.wm43=f(27); p.wpr=f(28);
    p.ws32=f(29); p.ws42=f(30); p.ws43=f(31); p.ws432=f(32); p.wsa=f(33); p.wst=f(34);
    p.x1=f(35); p.x2=f(36); p.x3=f(37); p.x4=f(38);
  }

  const int Bn = 32;
  const int HW44 = 44 * 44, HW22 = 22 * 22, HW88 = 88 * 88;
  const size_t S44 = (size_t)Bn * 32 * HW44;
  const size_t S22 = (size_t)Bn * 32 * HW22;
  const size_t S11 = (size_t)Bn * 32 * 121;
  const size_t S1_88 = (size_t)Bn * HW88;
  const size_t S64_88 = (size_t)Bn * 64 * HW88;
  const size_t S32_88 = (size_t)Bn * 32 * HW88;
  const size_t S16_44 = (size_t)Bn * 16 * HW44;
  const size_t S1_44 = (size_t)Bn * HW44;

  float* ws = (float*)d_ws;
  size_t off = 0;
  auto A = [&](size_t n) { float* r = ws + off; off += n; return r; };
  float *f2=A(S44), *f3=A(S22), *f4=A(S11);
  float *u3=A(S44), *u4=A(S22), *uu4=A(S44);
  float *tmpA=A(S44), *tmp22=A(S22);
  float *x32=A(S44), *x42=A(S44), *x43=A(S22);
  float *upx43=A(S44), *x432=A(S44);
  float *cat96=A(3 * S44), *o43=A(S22), *ba=A(S44);
  float *avg=A(2048), *mxv=A(2048), *attn=A(2048);
  float *sa2=A(2 * S1_88), *spat=A(S1_88);
  float *ta=A(S64_88), *sa88=A(S32_88), *sa44=A(S44);
  float *edge=A(S1_44);
  float *xstate=A(S16_44), *xproj=A(S16_44), *xmask=A(S16_44);
  float *anchor=A(8192), *proj=A(S16_44), *xbb=A(8192), *xb2=A(8192);
  float *xsb=A(S16_44);
  (void)ws_size; (void)out_size;

  auto G = [](size_t n) { return dim3((unsigned)((n + 255) / 256)); };

  // backbone feature projections (1x1 WMMA GEMMs; B staged by TDM)
  launch_conv(stream, p.x2, p.w2, p.b2, nullptr, f2, Bn, 128, 32, 44, 44, 1, 0, 1);
  launch_conv(stream, p.x3, p.w3, p.b3, nullptr, f3, Bn, 320, 32, 22, 22, 1, 0, 1);
  launch_conv(stream, p.x4, p.w4, p.b4, nullptr, f4, Bn, 512, 32, 11, 11, 1, 0, 1);
  k_resize<<<G(S44), 256, 0, stream>>>(f3, u3, Bn, 32, 22, 22, 44, 44, 32, 0);
  k_resize<<<G(S22), 256, 0, stream>>>(f4, u4, Bn, 32, 11, 11, 22, 22, 32, 0);
  k_resize<<<G(S44), 256, 0, stream>>>(u4, uu4, Bn, 32, 22, 22, 44, 44, 32, 0);

  // cross-scale difference branches (3x3 WMMA convs)
  k_absdiff<<<G(S44), 256, 0, stream>>>(u3, f2, tmpA, (int)S44);
  launch_conv(stream, tmpA, p.ws32, p.bs32, nullptr, x32, Bn, 32, 32, 44, 44, 3, 1, 1);
  k_absdiff<<<G(S44), 256, 0, stream>>>(uu4, f2, tmpA, (int)S44);
  launch_conv(stream, tmpA, p.ws42, p.bs42, nullptr, x42, Bn, 32, 32, 44, 44, 3, 1, 1);
  k_absdiff<<<G(S22), 256, 0, stream>>>(u4, f3, tmp22, (int)S22);
  launch_conv(stream, tmp22, p.ws43, p.bs43, nullptr, x43, Bn, 32, 32, 22, 22, 3, 1, 1);
  k_resize<<<G(S44), 256, 0, stream>>>(x43, upx43, Bn, 32, 22, 22, 44, 44, 32, 0);
  k_add3<<<G(S44), 256, 0, stream>>>(x32, x42, upx43, tmpA, (int)S44);
  launch_conv(stream, tmpA, p.ws432, p.bs432, nullptr, x432, Bn, 32, 32, 44, 44, 3, 1, 1);

  // modulation branches -> concat buffer [up(o4_3) | o4_2 | o3_2]
  launch_conv(stream, u3, p.wm32, p.bm32, nullptr, tmpA, Bn, 32, 32, 44, 44, 3, 1, 1);
  k_mul3_off<<<G(S44), 256, 0, stream>>>(tmpA, f2, x32, cat96, Bn, 32, HW44, 96, 64);
  launch_conv(stream, uu4, p.wm42, p.bm42, nullptr, tmpA, Bn, 32, 32, 44, 44, 3, 1, 1);
  k_mul3_off<<<G(S44), 256, 0, stream>>>(tmpA, f2, x42, cat96, Bn, 32, HW44, 96, 32);
  launch_conv(stream, u4, p.wm43, p.bm43, nullptr, tmp22, Bn, 32, 32, 22, 22, 3, 1, 1);
  k_mul3_off<<<G(S22), 256, 0, stream>>>(tmp22, f3, x43, o43, Bn, 32, HW22, 32, 0);
  k_resize<<<G(S44), 256, 0, stream>>>(o43, cat96, Bn, 32, 22, 22, 44, 44, 96, 0);

  launch_conv(stream, cat96, p.w5, p.b5, nullptr, tmpA, Bn, 96, 32, 44, 44, 3, 1, 1);
  k_bafuse<<<G(S44), 256, 0, stream>>>(tmpA, x432, f2, u3, uu4, ba, (int)S44);

  // SE channel attention + CBAM spatial attention on x1
  k_se_reduce<<<dim3(Bn * 64), 256, 0, stream>>>(p.x1, avg, mxv, HW88);
  k_se_attn<<<dim3(Bn), 64, 0, stream>>>(avg, mxv, p.wfc1, p.wfc2, attn);
  k_spat_mm<<<G(S1_88), 256, 0, stream>>>(p.x1, attn, sa2, HW88);
  k_conv7_sig<<<G(S1_88), 256, 0, stream>>>(sa2, p.wsa, spat, 88, 88);
  k_ta<<<G(S64_88), 256, 0, stream>>>(p.x1, attn, spat, ta, HW88);
  launch_conv(stream, ta, p.w1, p.b1, nullptr, sa88, Bn, 64, 32, 88, 88, 1, 0, 0);
  k_resize<<<G(S44), 256, 0, stream>>>(sa88, sa44, Bn, 32, 88, 88, 44, 44, 32, 0);
  k_edge<<<G(S1_44), 256, 0, stream>>>(sa44, edge, HW44);

  // GloRe graph reasoning
  launch_conv(stream, ba, p.wst, p.bst, nullptr, xstate, Bn, 32, 16, 44, 44, 1, 0, 0);
  launch_conv(stream, ba, p.wpr, p.bpr, nullptr, xproj, Bn, 32, 16, 44, 44, 1, 0, 0);
  k_xmask<<<G(S16_44), 256, 0, stream>>>(xproj, edge, xmask, HW44);
  k_anchor<<<G(8192), 256, 0, stream>>>(xmask, anchor);
  k_proj<<<G(S1_44), 256, 0, stream>>>(anchor, xproj, proj, HW44);
  k_xb<<<dim3(Bn * 16), 256, 0, stream>>>(xstate, proj, xbb, HW44);
  k_gcn<<<dim3(Bn), 256, 0, stream>>>(xbb, p.wg1, p.bg1, p.wg2, xb2);
  k_xs<<<G(S1_44), 256, 0, stream>>>(xb2, proj, xsb, HW44);

  // final: out = ba + conv1x1(xs, wex)   (residual fused into WMMA epilogue)
  launch_conv(stream, xsb, p.wex, nullptr, ba, (float*)d_out, Bn, 16, 32, 44, 44, 1, 0, 0);
}
